// HybridQuantumClassifier_65481071410191
// MI455X (gfx1250) — compile-verified, matrix operand order checked by reading, not run on hardware
//
#include <hip/hip_runtime.h>
#include <hip/hip_bf16.h>

typedef __attribute__((ext_vector_type(16))) _Float16 v16h;
typedef __attribute__((ext_vector_type(8)))  _Float16 v8h;
typedef __attribute__((ext_vector_type(8)))  float    v8f;

// ---------------- problem dimensions ----------------
constexpr int BATCH = 128;
constexpr int IN_C = 3,  IN_H = 250, IN_W = 250;
constexpr int C1_OC = 6,  C1_H = 124, C1_W = 124;   // conv1: 5x5 s2 p1
constexpr int P1_H = 123, P1_W = 123;               // maxpool 2x2 s1
constexpr int C2_OC = 15, C2_H = 62,  C2_W = 62;    // conv2: 3x3 s2 p1
constexpr int P2_H = 61,  P2_W = 61;                // maxpool 2x2 s1
constexpr int KDIM = C2_OC * P2_H * P2_W;           // 55815
constexpr int KPAD = 55840;                         // next multiple of 32
constexpr int FC1_N = 120;
constexpr int NPAD = 128;
constexpr int KSTEPS = KPAD / 32;                   // 1745 wmma K-steps
constexpr int KC = 64;                              // K-chunks (4096 waves total)
constexpr int SPC = (KSTEPS + KC - 1) / KC;         // 28 steps per chunk

// ---------------- workspace layout (bytes) ----------------
constexpr size_t SZ_C1  = (size_t)BATCH * C1_OC * C1_H * C1_W * 4;  // 47.2 MB, reused by conv2 out
constexpr size_t SZ_P1  = (size_t)BATCH * C1_OC * P1_H * P1_W * 4;  // 46.5 MB, reused by Ah
constexpr size_t SZ_WH  = (size_t)NPAD * KPAD * 2;                  // 14.3 MB
constexpr size_t OFF_R0 = 0;                 // conv1 out  -> later conv2 out
constexpr size_t OFF_R1 = OFF_R0 + SZ_C1;    // pool1 out  -> later Ah (f16)
constexpr size_t OFF_WH = OFF_R1 + SZ_P1;    // Wh (f16)
constexpr size_t OFF_CA = OFF_WH + SZ_WH;    // Cacc (128x128 f32)

// ---------------- conv1 + relu ----------------
__global__ void conv1_relu_kernel(const float* __restrict__ x,
                                  const float* __restrict__ w,
                                  const float* __restrict__ bias,
                                  float* __restrict__ out) {
  int idx = blockIdx.x * 256 + threadIdx.x;
  if (idx >= BATCH * C1_OC * C1_H * C1_W) return;
  int ox = idx % C1_W; int t = idx / C1_W;
  int oy = t % C1_H;   t /= C1_H;
  int oc = t % C1_OC;  int b = t / C1_OC;
  int iy0 = oy * 2 - 1, ix0 = ox * 2 - 1;
  float acc = bias[oc];
  const float* xb = x + (size_t)b * IN_C * IN_H * IN_W;
  const float* wk = w + oc * IN_C * 25;
  for (int ic = 0; ic < IN_C; ++ic)
    for (int ky = 0; ky < 5; ++ky) {
      int iy = iy0 + ky; if ((unsigned)iy >= (unsigned)IN_H) continue;
      const float* xr = xb + ((size_t)ic * IN_H + iy) * IN_W;
      const float* wr = wk + (ic * 5 + ky) * 5;
      for (int kx = 0; kx < 5; ++kx) {
        int ix = ix0 + kx; if ((unsigned)ix >= (unsigned)IN_W) continue;
        acc = fmaf(xr[ix], wr[kx], acc);
      }
    }
  out[idx] = fmaxf(acc, 0.f);
}

// ---------------- maxpool 2x2 s1 (conv1 -> pool1) ----------------
__global__ void pool1_kernel(const float* __restrict__ in, float* __restrict__ out) {
  int idx = blockIdx.x * 256 + threadIdx.x;
  if (idx >= BATCH * C1_OC * P1_H * P1_W) return;
  int x = idx % P1_W; int t = idx / P1_W;
  int y = t % P1_H;   t /= P1_H;
  const float* base = in + (((size_t)t * C1_H) + y) * C1_W + x;
  out[idx] = fmaxf(fmaxf(base[0], base[1]), fmaxf(base[C1_W], base[C1_W + 1]));
}

// ---------------- conv2 + relu ----------------
__global__ void conv2_relu_kernel(const float* __restrict__ in,
                                  const float* __restrict__ w,
                                  const float* __restrict__ bias,
                                  float* __restrict__ out) {
  int idx = blockIdx.x * 256 + threadIdx.x;
  if (idx >= BATCH * C2_OC * C2_H * C2_W) return;
  int ox = idx % C2_W; int t = idx / C2_W;
  int oy = t % C2_H;   t /= C2_H;
  int oc = t % C2_OC;  int b = t / C2_OC;
  int iy0 = oy * 2 - 1, ix0 = ox * 2 - 1;
  float acc = bias[oc];
  const float* xb = in + (size_t)b * C1_OC * P1_H * P1_W;
  const float* wk = w + oc * C1_OC * 9;
  for (int ic = 0; ic < C1_OC; ++ic)
    for (int ky = 0; ky < 3; ++ky) {
      int iy = iy0 + ky; if ((unsigned)iy >= (unsigned)P1_H) continue;
      const float* xr = xb + ((size_t)ic * P1_H + iy) * P1_W;
      const float* wr = wk + (ic * 3 + ky) * 3;
      for (int kx = 0; kx < 3; ++kx) {
        int ix = ix0 + kx; if ((unsigned)ix >= (unsigned)P1_W) continue;
        acc = fmaf(xr[ix], wr[kx], acc);
      }
    }
  out[idx] = fmaxf(acc, 0.f);
}

// ---------------- maxpool 2x2 s1 + f16 convert + K padding ----------------
__global__ void pool2_cvt_kernel(const float* __restrict__ c2, _Float16* __restrict__ Ah) {
  int idx = blockIdx.x * 256 + threadIdx.x;
  if (idx >= BATCH * KPAD) return;
  int k = idx % KPAD; int b = idx / KPAD;
  float v = 0.f;
  if (k < KDIM) {
    int c = k / (P2_H * P2_W); int rem = k % (P2_H * P2_W);
    int y = rem / P2_W;        int x = rem % P2_W;
    const float* base = c2 + (((size_t)(b * C2_OC + c) * C2_H) + y) * C2_W + x;
    v = fmaxf(fmaxf(base[0], base[1]), fmaxf(base[C2_W], base[C2_W + 1]));
  }
  Ah[idx] = (_Float16)v;
}

// ---------------- fc1 weight repack f32 -> f16, pad N to 128 and K to 55840 ----------------
__global__ void wcvt_kernel(const float* __restrict__ w, _Float16* __restrict__ Wh) {
  int idx = blockIdx.x * 256 + threadIdx.x;
  if (idx >= NPAD * KPAD) return;
  int k = idx % KPAD; int n = idx / KPAD;
  float v = (n < FC1_N && k < KDIM) ? w[(size_t)n * KDIM + k] : 0.f;
  Wh[idx] = (_Float16)v;
}

// ---------------- init accumulator with fc1 bias ----------------
__global__ void initc_kernel(const float* __restrict__ bias, float* __restrict__ C) {
  int idx = blockIdx.x * 256 + threadIdx.x;
  if (idx >= NPAD * NPAD) return;
  int n = idx & (NPAD - 1);
  C[idx] = (n < FC1_N) ? bias[n] : 0.f;
}

// ---------------- fc1 GEMM via v_wmma_f32_16x16x32_f16 ----------------
// 64 output tiles (8m x 8n) x KC K-chunks; one wave per (tile, chunk).
// K-loop unrolled x2 with two independent accumulators (two WMMA chains
// in flight), plus explicit global_prefetch ~512B ahead per operand stream.
// Fragment layouts per CDNA5 ISA 7.12.2 (16-bit A 16x32; C/D 16x16).
__global__ void fc1_wmma_kernel(const _Float16* __restrict__ Ah,
                                const _Float16* __restrict__ Wh,
                                float* __restrict__ Cacc) {
  int wave  = (blockIdx.x * blockDim.x + threadIdx.x) >> 5;
  int lane  = threadIdx.x & 31;
  int tile  = wave / KC;
  int chunk = wave % KC;
  int mt = tile >> 3, nt = tile & 7;
  int row = lane & 15;
  int kb  = (lane >> 4) * 8;                    // lane group selects K offset 0/8
  const _Float16* ap = Ah + (size_t)(mt * 16 + row) * KPAD + kb;
  const _Float16* bp = Wh + (size_t)(nt * 16 + row) * KPAD + kb;
  int k0 = chunk * SPC;
  int k1 = (k0 + SPC < KSTEPS) ? (k0 + SPC) : KSTEPS;
  if (k1 < k0) k1 = k0;
  v8f acc0 = {};
  v8f acc1 = {};

  auto ldfrag = [](const _Float16* p) -> v16h {
    v8h lo = *(const v8h*)(p);        // K = kb+0..7   (16B aligned)
    v8h hi = *(const v8h*)(p + 16);   // K = kb+16..23
    return __builtin_shufflevector(lo, hi, 0,1,2,3,4,5,6,7,8,9,10,11,12,13,14,15);
  };

  int kk = k0;
  for (; kk + 2 <= k1; kk += 2) {
    const _Float16* a0 = ap + (size_t)kk * 32;
    const _Float16* b0 = bp + (size_t)kk * 32;
    // keep both streams primed ~8 steps (512B) ahead
    __builtin_prefetch(a0 + 32 * 8, 0, 3);
    __builtin_prefetch(b0 + 32 * 8, 0, 3);
    v16h aA = ldfrag(a0);
    v16h bA = ldfrag(b0);
    v16h aB = ldfrag(a0 + 32);
    v16h bB = ldfrag(b0 + 32);
    acc0 = __builtin_amdgcn_wmma_f32_16x16x32_f16(false, aA, false, bA,
                                                  (short)0, acc0, false, false);
    acc1 = __builtin_amdgcn_wmma_f32_16x16x32_f16(false, aB, false, bB,
                                                  (short)0, acc1, false, false);
  }
  if (kk < k1) {  // odd tail (wave-uniform)
    v16h aA = ldfrag(ap + (size_t)kk * 32);
    v16h bA = ldfrag(bp + (size_t)kk * 32);
    acc0 = __builtin_amdgcn_wmma_f32_16x16x32_f16(false, aA, false, bA,
                                                  (short)0, acc0, false, false);
  }

  // C/D layout: lane holds column N = lane&15; VGPR r holds row M = r + 8*(lane>=16)
  int n     = nt * 16 + (lane & 15);
  int mbase = mt * 16 + ((lane >> 4) ? 8 : 0);
  #pragma unroll
  for (int r = 0; r < 8; ++r)
    atomicAdd(&Cacc[(size_t)(mbase + r) * NPAD + n], acc0[r] + acc1[r]);
}

// ---------------- tail: relu->fc2->relu->fc3->mlp->quantum ----------------
__global__ void tail_kernel(const float* __restrict__ Cacc,
                            const float* __restrict__ fc2_w, const float* __restrict__ fc2_b,
                            const float* __restrict__ fc3_w, const float* __restrict__ fc3_b,
                            const float* __restrict__ mlp1_w, const float* __restrict__ mlp1_b,
                            const float* __restrict__ mlp2_w, const float* __restrict__ mlp2_b,
                            float* __restrict__ out) {
  __shared__ float s_fc2w[84 * 120];
  int tid = threadIdx.x;
  for (int i = tid; i < 84 * 120; i += 128) s_fc2w[i] = fc2_w[i];
  __syncthreads();

  int b = tid;  // one thread per batch row (128 threads)
  float h[120];
  for (int j = 0; j < 120; ++j) h[j] = fmaxf(Cacc[(size_t)b * NPAD + j], 0.f);

  float y = fc3_b[0];
  for (int i = 0; i < 84; ++i) {
    float v = fc2_b[i];
    const float* wr = &s_fc2w[i * 120];
    for (int j = 0; j < 120; ++j) v = fmaf(wr[j], h[j], v);
    y = fmaf(fc3_w[i], fmaxf(v, 0.f), y);
  }

  float p[8];
  {
    float t[32];
    for (int r = 0; r < 32; ++r) t[r] = fmaxf(fmaf(mlp1_w[r], y, mlp1_b[r]), 0.f);
    for (int q = 0; q < 8; ++q) {
      float v = mlp2_b[q];
      for (int r = 0; r < 32; ++r) v = fmaf(mlp2_w[q * 32 + r], t[r], v);
      p[q] = v;
    }
  }

  // 4-qubit real statevector: RY layers + CNOT chain, 2 layers.
  float st[16];
  for (int i = 0; i < 16; ++i) st[i] = 0.f;
  st[0] = 1.f;
  for (int l = 0; l < 2; ++l) {
    for (int j = 0; j < 4; ++j) {
      float th = p[l * 4 + j];
      float c = cosf(0.5f * th), s = sinf(0.5f * th);
      int bit = 1 << (3 - j);
      for (int i = 0; i < 16; ++i)
        if (!(i & bit)) {
          float s0 = st[i], s1 = st[i | bit];
          st[i]       = c * s0 - s * s1;
          st[i | bit] = s * s0 + c * s1;
        }
    }
    for (int j = 0; j < 3; ++j) {
      int cb = 1 << (3 - j), tb = 1 << (2 - j);
      for (int i = 0; i < 16; ++i)
        if ((i & cb) && !(i & tb)) {
          float tmp = st[i]; st[i] = st[i | tb]; st[i | tb] = tmp;
        }
    }
  }
  float q0 = 0.f, q1 = 0.f;
  for (int i = 0; i < 8;  ++i) q0 += st[i] * st[i];
  for (int i = 8; i < 16; ++i) q1 += st[i] * st[i];
  out[b]       = q0 - q1;
  out[128 + b] = 1.f - (q0 - q1);
}

// ---------------- launcher ----------------
extern "C" void kernel_launch(void* const* d_in, const int* in_sizes, int n_in,
                              void* d_out, int out_size, void* d_ws, size_t ws_size,
                              hipStream_t stream) {
  const float* x       = (const float*)d_in[0];
  const float* conv1_w = (const float*)d_in[1];
  const float* conv1_b = (const float*)d_in[2];
  const float* conv2_w = (const float*)d_in[3];
  const float* conv2_b = (const float*)d_in[4];
  const float* fc1_w   = (const float*)d_in[5];
  const float* fc1_b   = (const float*)d_in[6];
  const float* fc2_w   = (const float*)d_in[7];
  const float* fc2_b   = (const float*)d_in[8];
  const float* fc3_w   = (const float*)d_in[9];
  const float* fc3_b   = (const float*)d_in[10];
  const float* mlp1_w  = (const float*)d_in[11];
  const float* mlp1_b  = (const float*)d_in[12];
  const float* mlp2_w  = (const float*)d_in[13];
  const float* mlp2_b  = (const float*)d_in[14];
  float* out = (float*)d_out;

  char* ws = (char*)d_ws;
  float*    conv1o = (float*)(ws + OFF_R0);
  float*    pool1o = (float*)(ws + OFF_R1);
  float*    conv2o = (float*)(ws + OFF_R0);   // aliases conv1o (dead by then)
  _Float16* Ah     = (_Float16*)(ws + OFF_R1);// aliases pool1o (dead by then)
  _Float16* Wh     = (_Float16*)(ws + OFF_WH);
  float*    Cacc   = (float*)(ws + OFF_CA);

  auto blocks = [](long n) { return (int)((n + 255) / 256); };

  // weight repack can go first (independent)
  wcvt_kernel<<<blocks((long)NPAD * KPAD), 256, 0, stream>>>(fc1_w, Wh);
  initc_kernel<<<blocks(NPAD * NPAD), 256, 0, stream>>>(fc1_b, Cacc);

  conv1_relu_kernel<<<blocks((long)BATCH * C1_OC * C1_H * C1_W), 256, 0, stream>>>(
      x, conv1_w, conv1_b, conv1o);
  pool1_kernel<<<blocks((long)BATCH * C1_OC * P1_H * P1_W), 256, 0, stream>>>(
      conv1o, pool1o);
  conv2_relu_kernel<<<blocks((long)BATCH * C2_OC * C2_H * C2_W), 256, 0, stream>>>(
      pool1o, conv2_w, conv2_b, conv2o);
  pool2_cvt_kernel<<<blocks((long)BATCH * KPAD), 256, 0, stream>>>(conv2o, Ah);

  // 64 tiles * 64 K-chunks = 4096 waves; 4 waves per 128-thread block
  fc1_wmma_kernel<<<(64 * KC) / 4, 128, 0, stream>>>(Ah, Wh, Cacc);

  tail_kernel<<<1, 128, 0, stream>>>(Cacc, fc2_w, fc2_b, fc3_w, fc3_b,
                                     mlp1_w, mlp1_b, mlp2_w, mlp2_b, out);
}
